// InfoNCELoss_53833120088584
// MI455X (gfx1250) — matrix operand order repaired and checked by reading, not compiled, not run on hardware
//
#include <hip/hip_runtime.h>
#include <math.h>

typedef __attribute__((ext_vector_type(16))) _Float16 v16h;
typedef __attribute__((ext_vector_type(8)))  _Float16 v8h;
typedef __attribute__((ext_vector_type(8)))  float    v8f;

#define B_N   16384
#define D_K   256
#define INV_T (1.0f / 0.07f)
#define EPSN  1e-12f

#define TILE_M 256   // q rows per block (2 strips of 16 per wave)
#define TILE_N 128   // d rows (columns) per block
#define KC     32
#define LDSW   40    // padded row width in halfs -> conflict-free strided reads

// ---------------------------------------------------------------------------
// Kernel 1: per-row L2 normalize q and d (fp32 math), emit f16 copies for the
// WMMA GEMM, compute the fp32 diagonal sim_ii/T, and zero the accumulators.
// One wave32 per row; lane i owns 8 contiguous floats.
// ---------------------------------------------------------------------------
__global__ void __launch_bounds__(256) norm_diag_kernel(
    const float* __restrict__ q, const float* __restrict__ d,
    _Float16* __restrict__ qh, _Float16* __restrict__ dh,
    float* __restrict__ diag, float* __restrict__ rowsum,
    float* __restrict__ colsum)
{
    const int tid  = threadIdx.x;
    const int gid  = blockIdx.x * 256 + tid;
    if (gid < B_N) {          // zero accumulators every call (graph replays)
        rowsum[gid] = 0.0f;
        colsum[gid] = 0.0f;
    }

    const int wave = tid >> 5;
    const int lane = tid & 31;
    const int row  = blockIdx.x * 8 + wave;

    const float* qr = q + (size_t)row * D_K + lane * 8;
    const float* dr = d + (size_t)row * D_K + lane * 8;

    float qv[8], dv[8];
    float ssq = 0.0f, ssd = 0.0f, dot = 0.0f;
#pragma unroll
    for (int i = 0; i < 8; ++i) {
        qv[i] = qr[i];
        dv[i] = dr[i];
        ssq += qv[i] * qv[i];
        ssd += dv[i] * dv[i];
        dot += qv[i] * dv[i];
    }
#pragma unroll
    for (int m = 16; m >= 1; m >>= 1) {
        ssq += __shfl_xor(ssq, m, 32);
        ssd += __shfl_xor(ssd, m, 32);
        dot += __shfl_xor(dot, m, 32);
    }
    const float iq = 1.0f / fmaxf(sqrtf(ssq), EPSN);
    const float id = 1.0f / fmaxf(sqrtf(ssd), EPSN);

    _Float16* qo = qh + (size_t)row * D_K + lane * 8;
    _Float16* dq = dh + (size_t)row * D_K + lane * 8;
#pragma unroll
    for (int i = 0; i < 8; ++i) {
        qo[i] = (_Float16)(qv[i] * iq);
        dq[i] = (_Float16)(dv[i] * id);
    }
    if (lane == 0) diag[row] = dot * iq * id * INV_T;
}

// ---------------------------------------------------------------------------
// Kernel 2: tiled WMMA GEMM with fused exp + row/col accumulation.
// Block = 256 threads (8 waves). Block tile = 256x128, K = 256 in 8 steps.
// Wave w owns two 16-row strips: rows w*16 and 128 + w*16 of the tile, across
// all 8 column subtiles -> every B fragment loaded from LDS feeds 2 WMMAs.
//
// Fragment layouts (CDNA5 ISA 7.12.2, wave32, 16-bit):
//  A (16x32, lane = M): lanes 0-15  -> K = kb+[0..7]  and kb+[16..23]
//                       lanes 16-31 -> K = kb+[8..15] and kb+[24..31]
//  B (32x16, lane = N): lanes 0-15  -> K = kb+[0..7]  and kb+[16..23]
//                       lanes 16-31 -> K = kb+[8..15] and kb+[24..31]
//  C (16x16): lanes 0-15 -> (M=v, N=lane); lanes 16-31 -> (M=v+8, N=lane-16)
// ---------------------------------------------------------------------------
__global__ void __launch_bounds__(256, 1) gemm_lse_kernel(
    const _Float16* __restrict__ qh, const _Float16* __restrict__ dh,
    float* __restrict__ rowsum, float* __restrict__ colsum)
{
    __shared__ _Float16 lq[TILE_M * LDSW];   // 20480 B
    __shared__ _Float16 ld[TILE_N * LDSW];   // 10240 B

    const int tid  = threadIdx.x;
    const int wave = tid >> 5;
    const int lane = tid & 31;
    const int rb   = blockIdx.y * TILE_M;  // query-row base
    const int cb   = blockIdx.x * TILE_N;  // database-row (column) base

    v8f acc[2][8] = {};

    // cooperative-load mapping: thread moves 16 halfs per 128-row panel
    const int lrow = tid >> 1;          // 0..127
    const int lcol = (tid & 1) * 16;    // 0 or 16

    // fragment mapping
    const int rA  = lane & 15;
    const int koA = (lane >> 4) * 8;    // A: 8-half chunks at koA and koA+16
    const int koB = (lane >> 4) * 8;    // B: 8-half chunks at koB and koB+16

    for (int kb = 0; kb < D_K; kb += KC) {
        const v8h* gq0 = (const v8h*)(qh + (size_t)(rb + lrow)       * D_K + kb + lcol);
        const v8h* gq1 = (const v8h*)(qh + (size_t)(rb + 128 + lrow) * D_K + kb + lcol);
        const v8h* gd  = (const v8h*)(dh + (size_t)(cb + lrow)       * D_K + kb + lcol);
        const v8h q0 = gq0[0], q1 = gq0[1];
        const v8h q2 = gq1[0], q3 = gq1[1];
        const v8h d0 = gd[0],  d1 = gd[1];

        __syncthreads();  // previous iteration's LDS reads complete
        *(v8h*)&lq[lrow * LDSW + lcol]             = q0;
        *(v8h*)&lq[lrow * LDSW + lcol + 8]         = q1;
        *(v8h*)&lq[(128 + lrow) * LDSW + lcol]     = q2;
        *(v8h*)&lq[(128 + lrow) * LDSW + lcol + 8] = q3;
        *(v8h*)&ld[lrow * LDSW + lcol]             = d0;
        *(v8h*)&ld[lrow * LDSW + lcol + 8]         = d1;
        __syncthreads();

        union { v16h v; v8h h[2]; } af0, af1;
        af0.h[0] = *(const v8h*)&lq[(wave * 16 + rA) * LDSW + koA];
        af0.h[1] = *(const v8h*)&lq[(wave * 16 + rA) * LDSW + koA + 16];
        af1.h[0] = *(const v8h*)&lq[(128 + wave * 16 + rA) * LDSW + koA];
        af1.h[1] = *(const v8h*)&lq[(128 + wave * 16 + rA) * LDSW + koA + 16];

#pragma unroll
        for (int t = 0; t < 8; ++t) {
            union { v16h v; v8h h[2]; } bf;
            bf.h[0] = *(const v8h*)&ld[(t * 16 + rA) * LDSW + koB];
            bf.h[1] = *(const v8h*)&ld[(t * 16 + rA) * LDSW + koB + 16];
            acc[0][t] = __builtin_amdgcn_wmma_f32_16x16x32_f16(
                false, af0.v, false, bf.v, (short)0, acc[0][t], false, false);
            acc[1][t] = __builtin_amdgcn_wmma_f32_16x16x32_f16(
                false, af1.v, false, bf.v, (short)0, acc[1][t], false, false);
        }
    }

    // Epilogue. |sim/T| <= 14.3 so exp is bounded (<=1.6e6); no max-shift.
    float rpart[2][8];
#pragma unroll
    for (int s = 0; s < 2; ++s)
#pragma unroll
        for (int v = 0; v < 8; ++v) rpart[s][v] = 0.0f;

#pragma unroll
    for (int t = 0; t < 8; ++t) {
        float cpart = 0.0f;
#pragma unroll
        for (int s = 0; s < 2; ++s) {
#pragma unroll
            for (int v = 0; v < 8; ++v) {
                const float e = __expf(acc[s][t][v] * INV_T);
                rpart[s][v] += e;   // row partial (columns of subtile t)
                cpart       += e;   // this lane's column, both strips
            }
        }
        cpart += __shfl_xor(cpart, 16, 32);  // merge the two half-wave copies
        if (lane < 16) atomicAdd(&colsum[cb + t * 16 + lane], cpart);
    }

#pragma unroll
    for (int s = 0; s < 2; ++s) {
#pragma unroll
        for (int v = 0; v < 8; ++v) {
            float r = rpart[s][v];
            r += __shfl_xor(r, 1, 32);
            r += __shfl_xor(r, 2, 32);
            r += __shfl_xor(r, 4, 32);
            r += __shfl_xor(r, 8, 32);  // lanes 0 / 16 hold the half-sums
            const int rbase = rb + s * 128 + wave * 16;
            if (lane == 0)  atomicAdd(&rowsum[rbase + v],     r);
            if (lane == 16) atomicAdd(&rowsum[rbase + 8 + v], r);
        }
    }
}

// ---------------------------------------------------------------------------
// Kernel 3: loss = (sum(log rowsum) + sum(log colsum) - 2*sum(diag)) / (2B)
// ---------------------------------------------------------------------------
__global__ void __launch_bounds__(256) finalize_kernel(
    const float* __restrict__ rowsum, const float* __restrict__ colsum,
    const float* __restrict__ diag, float* __restrict__ out)
{
    __shared__ float wsum[8];
    const int tid = threadIdx.x;
    float acc = 0.0f;
    for (int i = tid; i < B_N; i += 256)
        acc += logf(rowsum[i]) + logf(colsum[i]) - 2.0f * diag[i];
#pragma unroll
    for (int m = 16; m >= 1; m >>= 1) acc += __shfl_xor(acc, m, 32);
    if ((tid & 31) == 0) wsum[tid >> 5] = acc;
    __syncthreads();
    if (tid == 0) {
        float s = 0.0f;
#pragma unroll
        for (int i = 0; i < 8; ++i) s += wsum[i];
        out[0] = s / (2.0f * (float)B_N);
    }
}

// ---------------------------------------------------------------------------
extern "C" void kernel_launch(void* const* d_in, const int* in_sizes, int n_in,
                              void* d_out, int out_size, void* d_ws, size_t ws_size,
                              hipStream_t stream) {
    const float* q = (const float*)d_in[0];
    const float* d = (const float*)d_in[1];

    char* ws = (char*)d_ws;
    _Float16* qh     = (_Float16*)ws;                                   // 8 MB
    _Float16* dh     = (_Float16*)(ws + (size_t)B_N * D_K * 2);         // 8 MB
    float*    rowsum = (float*)  (ws + (size_t)B_N * D_K * 4);          // 64 KB
    float*    colsum = rowsum + B_N;                                    // 64 KB
    float*    diag   = colsum + B_N;                                    // 64 KB

    norm_diag_kernel<<<B_N / 8, 256, 0, stream>>>(q, d, qh, dh, diag, rowsum, colsum);

    dim3 grid(B_N / TILE_N, B_N / TILE_M);
    gemm_lse_kernel<<<grid, 256, 0, stream>>>(qh, dh, rowsum, colsum);

    finalize_kernel<<<1, 256, 0, stream>>>(rowsum, colsum, diag, (float*)d_out);
}
